// ContextShadowReasoning_47871705481634
// MI455X (gfx1250) — matrix operand order repaired and checked by reading, not compiled, not run on hardware
//
#include <hip/hip_runtime.h>
#include <hip/hip_bf16.h>

typedef _Float16 f16;
typedef __attribute__((ext_vector_type(16))) _Float16 v16h;
typedef __attribute__((ext_vector_type(8)))  _Float16 v8h;
typedef __attribute__((ext_vector_type(8)))  float    v8f;

#define HW 4096   // 64*64
#define IMG 64

static __device__ __forceinline__ v8f wmma_f16(v16h a, v16h b, v8f c) {
  // D = A(16x32 f16) * B(32x16 f16) + C(16x16 f32)
  return __builtin_amdgcn_wmma_f32_16x16x32_f16(false, a, false, b, (short)0, c, false, false);
}

static __device__ __forceinline__ v16h pack16(v8h lo, v8h hi) {
  v16h r;
#pragma unroll
  for (int i = 0; i < 8; ++i) { r[i] = lo[i]; r[i + 8] = hi[i]; }
  return r;
}

// ---------------------------------------------------------------------------
// f32 -> f16 conversion
// ---------------------------------------------------------------------------
__global__ void cvt_f32_f16(const float* __restrict__ in, f16* __restrict__ out, int n) {
  int i = blockIdx.x * blockDim.x + threadIdx.x;
  if (i < n) out[i] = (f16)in[i];
}

// ---------------------------------------------------------------------------
// Pack conv weights [Cout,Cin,KH,KW] f32 into per-lane WMMA B-fragment layout:
// pk[((ntile*numK + ks)*32 + lane)*16 + h], where
//   co  = ntile*16 + (lane&15)
//   tap = ks / (Cin/32), cc = ks % (Cin/32)
//   cin = cc*32 + (lane>>4)*8 + (h&7) + (h>>3)*16   (B-frag: K = (lane/16)*8 + (h&7) + (h/8)*16)
// ---------------------------------------------------------------------------
__global__ void pack_w(const float* __restrict__ w, f16* __restrict__ pk,
                       int Cout, int Cin, int KH, int KW) {
  long total = (long)Cout * Cin * KH * KW;
  long gid = (long)blockIdx.x * blockDim.x + threadIdx.x;
  if (gid >= total) return;
  int numK   = (KH * KW * Cin) >> 5;
  int h      = (int)(gid & 15);
  int lane   = (int)((gid >> 4) & 31);
  long rest  = gid >> 9;
  int ks     = (int)(rest % numK);
  int ntile  = (int)(rest / numK);
  int co     = ntile * 16 + (lane & 15);
  int cinCh  = Cin >> 5;
  int tap    = ks / cinCh;
  int cc     = ks % cinCh;
  int cin    = cc * 32 + (lane >> 4) * 8 + (h & 7) + (h >> 3) * 16;
  int kh     = tap / KW, kw = tap % KW;
  pk[gid] = (f16)w[(((long)co * Cin + cin) * KH + kh) * KW + kw];
}

// ---------------------------------------------------------------------------
// Implicit-GEMM convolution with WMMA.
//   M = pixels (one image row of 64 per block), N = output channels,
//   K = KH*KW*Cin in steps of 32 (fixed tap, 32 input channels).
// Block: 256 threads = 8 waves, 4(M) x 2(N); each wave carries NACC
// accumulators along N -> block tile = 64 pix x (32*NACC) co.
// Double-buffered LDS staging (2 x 8KB): next K-pair's global loads are
// issued before the current pair's WMMAs; ONE barrier per K-pair.
// grid = (64 rows, Cout/(32*NACC), B)
// Output: f16, either [b][co][4096] (coutT==0) or transposed [b][pix][coutT].
// ---------------------------------------------------------------------------
template <int NACC>
__global__ __launch_bounds__(256) void conv_wmma(
    const f16* __restrict__ xin, const f16* __restrict__ pk,
    const float* __restrict__ bias, f16* __restrict__ outp,
    int Cin, int KH, int KW, int pad, int dil,
    long bsIn, long bsOut, int coutT) {
  __shared__ alignas(16) f16 smem[2 * IMG * 64];  // 2 x [pixel 0..63][cin 0..63]

  const int t    = threadIdx.x;
  const int lane = t & 31;
  const int wv   = t >> 5;
  const int wn   = wv >> 2;          // 0..1  (N direction)
  const int wm   = wv & 3;           // 0..3  (M direction)
  const int hrow = blockIdx.x;       // image row
  const int b    = blockIdx.z;
  const int cinCh = Cin >> 5;
  const int numK  = KH * KW * cinCh;
  const int ntileBase = blockIdx.y * (2 * NACC) + wn * NACC;

  v8f acc[NACC];
  v8f zero = {};
#pragma unroll
  for (int j = 0; j < NACC; ++j) acc[j] = zero;

  const int px    = t & 63;
  const int cbase = t >> 6;                    // 0..3
  const f16* xb = xin + (long)b * bsIn;

  // stage one K-pair (2 x 64pix x 32cin) into dst, coalesced over pixels
  auto stage_pair = [&](f16* dst, int ks0) {
#pragma unroll
    for (int u = 0; u < 2; ++u) {
      const int ks  = ks0 + u;
      const int tap = ks / cinCh, cc = ks % cinCh;
      const int kh  = tap / KW,  kw = tap % KW;
      const int y    = hrow + kh * dil - pad;
      const int xoff = kw * dil - pad;
      const int xc   = px + xoff;
      const bool ok  = (y >= 0) && (y < IMG) && (xc >= 0) && (xc < IMG);
      const long base = (long)(cc * 32) * HW + (long)y * IMG + xc;
      // speculative prefetch of the slab after next (dropped if OOB)
      __builtin_prefetch(xb + base + (long)64 * HW, 0, 1);
#pragma unroll
      for (int i = 0; i < 8; ++i) {
        const int ci = cbase + i * 4;          // 0..31
        f16 vL = (f16)0.0f;
        if (ok) vL = xb[base + (long)ci * HW];
        dst[px * 64 + u * 32 + ci] = vL;
      }
    }
  };

  f16* buf0 = smem;
  f16* buf1 = smem + IMG * 64;

  // numK is even for all shapes used here (392, 200, 16, 8).
  stage_pair(buf0, 0);
  __syncthreads();

  const int p  = wm * 16 + (lane & 15);
  const int ab = (lane >> 4) * 8;

  for (int ks0 = 0; ks0 < numK; ks0 += 2) {
    f16* cur = ((ks0 >> 1) & 1) ? buf1 : buf0;
    f16* nxt = ((ks0 >> 1) & 1) ? buf0 : buf1;
    if (ks0 + 2 < numK) stage_pair(nxt, ks0 + 2);  // overlap with WMMAs below

    // ---- 2 K-steps x NACC WMMAs from 'cur' ----
#pragma unroll
    for (int u = 0; u < 2; ++u) {
      const v8h lo = *(const v8h*)&cur[p * 64 + u * 32 + ab];
      const v8h hi = *(const v8h*)&cur[p * 64 + u * 32 + ab + 16];
      const v16h af = pack16(lo, hi);
      const int ks = ks0 + u;
#pragma unroll
      for (int j = 0; j < NACC; ++j) {
        const f16* bp = pk + (((long)(ntileBase + j) * numK + ks) * 32 + lane) * 16;
        const v16h bf = *(const v16h*)bp;
        acc[j] = wmma_f16(af, bf, acc[j]);
      }
    }
    __syncthreads();   // fences: 'nxt' writes done; 'cur' reads done
  }

  // ---- epilogue: bias add, store f16 ----
#pragma unroll
  for (int j = 0; j < NACC; ++j) {
    const int co = (ntileBase + j) * 16 + (lane & 15);
    const float bv = bias[co];
    if (coutT == 0) {
      // lane's 8 rows are 8 consecutive pixels at fixed co -> one b128 store
      v8h ov;
#pragma unroll
      for (int r = 0; r < 8; ++r) ov[r] = (f16)(acc[j][r] + bv);
      const int pgBase = hrow * IMG + wm * 16 + (lane >> 4) * 8;
      *(v8h*)&outp[(long)b * bsOut + (long)co * HW + pgBase] = ov;
    } else {
#pragma unroll
      for (int r = 0; r < 8; ++r) {
        const int pg = hrow * IMG + wm * 16 + r + (lane >> 4) * 8;
        outp[((long)b * HW + pg) * coutT + co] = (f16)(acc[j][r] + bv);
      }
    }
  }
}

// ---------------------------------------------------------------------------
// Flash attention (transposed formulation) + mask epilogue.
//   S^T[n,m] = sum_c k[c,n] q[c,m]   (A = K^T rows=keys, B = Q cols=queries)
//   O^T[c,m] = sum_n v[c,n] P^T[n,m] (A = V rows=c,      B = P^T cols=queries)
// Each lane owns ONE query column m = lane&15: softmax max/sum are
// within-lane reductions + a single shfl_xor(16); running stats are per-lane
// scalars and the accumulator rescale is a uniform vector-scalar multiply.
// qT,kT: [b][4096][32] f16 (transposed). vmat: [b][256][4096] f16.
// Each wave owns 16 query columns; block = 8 waves = 128; grid = (32, B).
// ---------------------------------------------------------------------------
__global__ __launch_bounds__(256) void attn_flash(
    const f16* __restrict__ qT, const f16* __restrict__ kT,
    const f16* __restrict__ vmat, const float* __restrict__ mask,
    float* __restrict__ out) {
  __shared__ alignas(16) f16 pbuf[8][16][32];    // wave-private P^T tiles [m][n]

  const int lane  = threadIdx.x & 31;
  const int wv    = threadIdx.x >> 5;
  const int hf    = lane >> 4;                   // 0/1 half of wave
  const int l16   = lane & 15;
  const int ab    = hf * 8;                      // A-frag K-base / D-row base
  const int b     = blockIdx.y;
  const int mtile = (blockIdx.x * 8 + wv) * 16;
  const int m     = mtile + l16;                 // this lane's query

  // --- Q as B-fragment: col m = l16, halves c = hf*8 + h (+16 for h>=8) ---
  const f16* qrow = qT + (long)b * HW * 32 + (long)m * 32;
  const v16h qf = pack16(*(const v8h*)(qrow + hf * 8),
                         *(const v8h*)(qrow + hf * 8 + 16));

  v8f o[16];                                     // O^T: 16 c-tiles x (8c x 16m)
  v8f zero = {};
#pragma unroll
  for (int i = 0; i < 16; ++i) o[i] = zero;
  float rm = -3.0e38f, rs = 0.0f;                // per-lane (per-query) stats

  const f16* kb = kT + (long)b * HW * 32;
  const f16* vb = vmat + (long)b * 256 * HW;

  for (int n0 = 0; n0 < HW; n0 += 32) {
    // ---- S^T tiles: A = K^T (rows n), B = Q ----
    const f16* kr0 = kb + (long)(n0 + l16) * 32;
    const f16* kr1 = kb + (long)(n0 + 16 + l16) * 32;
    __builtin_prefetch(kr0 + 32 * 32, 0, 1);     // next key tile
    const v16h ka0 = pack16(*(const v8h*)(kr0 + ab), *(const v8h*)(kr0 + ab + 16));
    const v16h ka1 = pack16(*(const v8h*)(kr1 + ab), *(const v8h*)(kr1 + ab + 16));
    v8f s0 = wmma_f16(ka0, qf, zero);            // keys n0+ab..+7   (per lane)
    v8f s1 = wmma_f16(ka1, qf, zero);            // keys n0+16+ab..+7

    // ---- per-lane online softmax for query m ----
    float mx = s0[0];
#pragma unroll
    for (int r = 1; r < 8; ++r) mx = fmaxf(mx, s0[r]);
#pragma unroll
    for (int r = 0; r < 8; ++r) mx = fmaxf(mx, s1[r]);
    mx = fmaxf(mx, __shfl_xor(mx, 16));          // combine two halves
    const float nm = fmaxf(rm, mx);
    const float sc = __expf(rm - nm);
    float p0[8], p1[8], ss = 0.0f;
#pragma unroll
    for (int r = 0; r < 8; ++r) {
      p0[r] = __expf(s0[r] - nm);
      p1[r] = __expf(s1[r] - nm);
      ss += p0[r] + p1[r];
    }
    ss += __shfl_xor(ss, 16);
    rs = rs * sc + ss;
    rm = nm;
#pragma unroll
    for (int cc = 0; cc < 16; ++cc) o[cc] *= sc; // uniform per-lane rescale

    // ---- bounce P^T through wave-private LDS: pbuf[m][n] ----
#pragma unroll
    for (int r = 0; r < 8; ++r) {
      pbuf[wv][l16][r + ab]      = (f16)p0[r];
      pbuf[wv][l16][16 + r + ab] = (f16)p1[r];
    }
    // B-fragment of P^T: col m = l16, halves n = hf*8 + h (+16 for h>=8)
    const f16* pr = &pbuf[wv][l16][hf * 8];
    const v16h pf = pack16(*(const v8h*)pr, *(const v8h*)(pr + 16));

    // ---- O^T[c,m] += V(rows c, K=n) * P^T ----
#pragma unroll
    for (int cc = 0; cc < 16; ++cc) {
      const f16* vr = vb + (long)(cc * 16 + l16) * HW + n0;
      __builtin_prefetch(vr + 32, 0, 1);         // next value tile
      const v16h vf = pack16(*(const v8h*)(vr + ab), *(const v8h*)(vr + ab + 16));
      o[cc] = wmma_f16(vf, pf, o[cc]);
    }
  }

  // ---- epilogue: normalize, apply (1 + mask_resized), store f32 ----
  const float inv = 1.0f / rs;
  const float mk  = 1.0f +
      mask[(long)b * 256 * 256 + (long)((m >> 6) * 4) * 256 + (m & 63) * 4];
  const float f = inv * mk;
#pragma unroll
  for (int cc = 0; cc < 16; ++cc) {
#pragma unroll
    for (int r = 0; r < 8; ++r) {
      const int c = cc * 16 + r + ab;
      out[((long)(b * 256 + c)) * HW + m] = o[cc][r] * f;
    }
  }
}

// ---------------------------------------------------------------------------
// Host-side orchestration
// ---------------------------------------------------------------------------
extern "C" void kernel_launch(void* const* d_in, const int* in_sizes, int n_in,
                              void* d_out, int out_size, void* d_ws, size_t ws_size,
                              hipStream_t stream) {
  (void)in_sizes; (void)n_in; (void)out_size; (void)ws_size;
  const float* x      = (const float*)d_in[0];
  const float* mask   = (const float*)d_in[1];
  const float* w_ctx  = (const float*)d_in[2];
  const float* b_ctx  = (const float*)d_in[3];
  const float* w_wide = (const float*)d_in[4];
  const float* b_wide = (const float*)d_in[5];
  const float* w_comb = (const float*)d_in[6];
  const float* b_comb = (const float*)d_in[7];
  const float* w_q    = (const float*)d_in[8];
  const float* b_q    = (const float*)d_in[9];
  const float* w_k    = (const float*)d_in[10];
  const float* b_k    = (const float*)d_in[11];
  const float* w_v    = (const float*)d_in[12];
  const float* b_v    = (const float*)d_in[13];
  float* out = (float*)d_out;

  // carve workspace (f16 buffers, 256B-aligned)
  size_t off = 0;
  auto alloc = [&](size_t nhalf) -> f16* {
    f16* p = (f16*)((char*)d_ws + off);
    off += ((nhalf * sizeof(f16) + 255) / 256) * 256;
    return p;
  };
  const int B = 4, C = 256;
  f16* xh      = alloc((size_t)B * C * HW);        // 4.19M
  f16* pk_ctx  = alloc((size_t)C * C * 49);
  f16* pk_wide = alloc((size_t)C * C * 25);
  f16* pk_comb = alloc((size_t)C * 2 * C);
  f16* pk_q    = alloc((size_t)32 * C);
  f16* pk_k    = alloc((size_t)32 * C);
  f16* pk_v    = alloc((size_t)C * C);
  f16* cat     = alloc((size_t)B * 2 * C * HW);    // [local ; wide]
  f16* multi   = alloc((size_t)B * C * HW);
  f16* qTb     = alloc((size_t)B * HW * 32);
  f16* kTb     = alloc((size_t)B * HW * 32);
  f16* vbuf    = alloc((size_t)B * C * HW);

  // 1) convert x to f16
  {
    int n = B * C * HW;
    cvt_f32_f16<<<(n + 255) / 256, 256, 0, stream>>>(x, xh, n);
  }
  // 2) pack all weights into WMMA B-fragment layout
  {
    long n;
    n = (long)C * C * 49; pack_w<<<(n + 255) / 256, 256, 0, stream>>>(w_ctx,  pk_ctx,  C, C, 7, 7);
    n = (long)C * C * 25; pack_w<<<(n + 255) / 256, 256, 0, stream>>>(w_wide, pk_wide, C, C, 5, 5);
    n = (long)C * 2 * C;  pack_w<<<(n + 255) / 256, 256, 0, stream>>>(w_comb, pk_comb, C, 2 * C, 1, 1);
    n = (long)32 * C;     pack_w<<<(n + 255) / 256, 256, 0, stream>>>(w_q,    pk_q,    32, C, 1, 1);
    n = (long)32 * C;     pack_w<<<(n + 255) / 256, 256, 0, stream>>>(w_k,    pk_k,    32, C, 1, 1);
    n = (long)C * C;      pack_w<<<(n + 255) / 256, 256, 0, stream>>>(w_v,    pk_v,    C, C, 1, 1);
  }
  // 3) local_ctx = conv7x7(x) -> cat[:, 0:256]   (block: 64pix x 128co)
  conv_wmma<4><<<dim3(64, 2, B), 256, 0, stream>>>(
      xh, pk_ctx, b_ctx, cat,
      C, 7, 7, 3, 1, (long)C * HW, (long)2 * C * HW, 0);
  // 4) wide_ctx = conv5x5 dil3(x) -> cat[:, 256:512]
  conv_wmma<4><<<dim3(64, 2, B), 256, 0, stream>>>(
      xh, pk_wide, b_wide, cat + (size_t)C * HW,
      C, 5, 5, 6, 3, (long)C * HW, (long)2 * C * HW, 0);
  // 5) multi = conv1x1(cat, 512->256)
  conv_wmma<4><<<dim3(64, 2, B), 256, 0, stream>>>(
      cat, pk_comb, b_comb, multi,
      2 * C, 1, 1, 0, 1, (long)2 * C * HW, (long)C * HW, 0);
  // 6) q = conv1x1(multi, 256->32), transposed [pix][32]
  conv_wmma<1><<<dim3(64, 1, B), 256, 0, stream>>>(
      multi, pk_q, b_q, qTb,
      C, 1, 1, 0, 1, (long)C * HW, 0, 32);
  // 7) k likewise
  conv_wmma<1><<<dim3(64, 1, B), 256, 0, stream>>>(
      multi, pk_k, b_k, kTb,
      C, 1, 1, 0, 1, (long)C * HW, 0, 32);
  // 8) v = conv1x1(multi, 256->256), layout [c][pix]
  conv_wmma<4><<<dim3(64, 2, B), 256, 0, stream>>>(
      multi, pk_v, b_v, vbuf,
      C, 1, 1, 0, 1, (long)C * HW, (long)C * HW, 0);
  // 9) flash attention + mask epilogue -> d_out (f32)
  attn_flash<<<dim3(32, B), 256, 0, stream>>>(qTb, kTb, vbuf, mask, out);
}
